// MultiHeadAttention_46334107189709
// MI455X (gfx1250) — matrix operand order repaired and checked by reading, not compiled, not run on hardware
//
#include <hip/hip_runtime.h>
#include <hip/hip_bf16.h>

// MI455X / gfx1250, wave32. Matmuls via v_wmma_f32_16x16x32_f16.
// Double-buffered async global->LDS staging (ASYNCcnt) overlapped with WMMA.

typedef __attribute__((ext_vector_type(16))) _Float16 v16h;
typedef __attribute__((ext_vector_type(8)))  _Float16 v8h;
typedef __attribute__((ext_vector_type(8)))  float    v8f;
typedef __attribute__((ext_vector_type(4)))  float    f4;
typedef __attribute__((ext_vector_type(4)))  int      v4i;

// v4i in specific address spaces for the async-LDS builtin
typedef __attribute__((address_space(1))) v4i v4i_g;   // global
typedef __attribute__((address_space(3))) v4i v4i_l;   // LDS

#define D_MODEL  2048
#define HEADS    16
#define HEAD_DIM 128
#define BATCH    2
#define SEQ      2048
#define MROWS    (BATCH * SEQ)   // 4096 flattened rows

// A-fragment (16-bit 16x32, ISA 7.12.2): per lane two contiguous runs of 8
// at K = 8*hh and 16+8*hh.  B-fragment (32x16): same K order as A.

__device__ __forceinline__ v8f wmma_f16(v16h a, v16h b, v8f c) {
    return __builtin_amdgcn_wmma_f32_16x16x32_f16(false, a, false, b, (short)0, c,
                                                  false, false);
}

__device__ __forceinline__ v16h combine(v8h lo, v8h hi) {
    v16h r;
#pragma unroll
    for (int i = 0; i < 8; ++i) { r[i] = lo[i]; r[8 + i] = hi[i]; }
    return r;
}

__device__ __forceinline__ v8h cvt_v8h(f4 a, f4 b) {
    v8h r;
#pragma unroll
    for (int i = 0; i < 4; ++i) { r[i] = (_Float16)a[i]; r[4 + i] = (_Float16)b[i]; }
    return r;
}

// --- async global -> LDS copy (16B per lane), ASYNCcnt tracked ----------------
#if __has_builtin(__builtin_amdgcn_global_load_async_to_lds_b128)
#define ASYNC_LDS 1
__device__ __forceinline__ void async_cp16(const _Float16* g, _Float16* l) {
    __builtin_amdgcn_global_load_async_to_lds_b128((v4i_g*)g, (v4i_l*)l, 0, 0);
}
#else
#define ASYNC_LDS 0
__device__ __forceinline__ void async_cp16(const _Float16* g, _Float16* l) {
    *(v8h*)l = *(const v8h*)g;     // fallback: synchronous 128-bit copy
}
#endif
__device__ __forceinline__ void async_wait0() {
#if ASYNC_LDS
#if __has_builtin(__builtin_amdgcn_s_wait_asynccnt)
    __builtin_amdgcn_s_wait_asynccnt(0);
#else
    asm volatile("s_wait_asynccnt 0" ::: "memory");
#endif
#endif
}

// --- DPP 16-lane butterfly reductions (inside each DPP row of 16 lanes) ------
template <int CTRL>
__device__ __forceinline__ float dpp_xfer(float x) {
    int v = __builtin_amdgcn_update_dpp(0, __builtin_bit_cast(int, x),
                                        CTRL, 0xF, 0xF, true);
    return __builtin_bit_cast(float, v);
}
__device__ __forceinline__ float row_max16(float x) {
    x = fmaxf(x, dpp_xfer<0xB1>(x));   // quad_perm(1,0,3,2)
    x = fmaxf(x, dpp_xfer<0x4E>(x));   // quad_perm(2,3,0,1)
    x = fmaxf(x, dpp_xfer<0x141>(x));  // row_half_mirror
    x = fmaxf(x, dpp_xfer<0x140>(x));  // row_mirror
    return x;
}
__device__ __forceinline__ float row_sum16(float x) {
    x += dpp_xfer<0xB1>(x);
    x += dpp_xfer<0x4E>(x);
    x += dpp_xfer<0x141>(x);
    x += dpp_xfer<0x140>(x);
    return x;
}

// ---------------------------------------------------------------------------
// Tiled transpose W[K][N] (fp32) -> Wt[N][K] (f16).
// ---------------------------------------------------------------------------
__global__ __launch_bounds__(256) void wtrans_kernel(
    const float* __restrict__ W, _Float16* __restrict__ Wt, int K, int N)
{
    __shared__ _Float16 tile[32][33];
    const int bn = blockIdx.x * 32;
    const int bk = blockIdx.y * 32;
    const int tid = threadIdx.x;
#pragma unroll
    for (int t = 0; t < 4; ++t) {
        int idx = tid + t * 256;
        int r = idx >> 5, c = idx & 31;         // r: k, c: n
        tile[c][r] = (_Float16)W[(size_t)(bk + r) * N + bn + c];
    }
    __syncthreads();
#pragma unroll
    for (int t = 0; t < 4; ++t) {
        int idx = tid + t * 256;
        int r = idx >> 5, c = idx & 31;         // r: n, c: k
        Wt[(size_t)(bn + r) * K + bk + c] = tile[r][c];
    }
}

// ---------------------------------------------------------------------------
// Per-head transpose: Vp[b][s][h*128+d] (f16) -> Vt[b][h][d][s].
// ---------------------------------------------------------------------------
__global__ __launch_bounds__(256) void vtrans_kernel(
    const _Float16* __restrict__ Vp, _Float16* __restrict__ Vt)
{
    __shared__ _Float16 tile[32][33];
    const int b  = blockIdx.z;
    const int s0 = blockIdx.x * 32;
    const int d0 = blockIdx.y * 32;
    const int tid = threadIdx.x;
#pragma unroll
    for (int t = 0; t < 4; ++t) {
        int idx = tid + t * 256;
        int r = idx >> 5, c = idx & 31;         // r: s, c: d
        tile[c][r] = Vp[((size_t)b * SEQ + s0 + r) * D_MODEL + d0 + c];
    }
    __syncthreads();
    const int h   = d0 >> 7;
    const int dl0 = d0 & 127;
#pragma unroll
    for (int t = 0; t < 4; ++t) {
        int idx = tid + t * 256;
        int r = idx >> 5, c = idx & 31;         // r: d, c: s
        Vt[(((size_t)b * HEADS + h) * HEAD_DIM + dl0 + r) * SEQ + s0 + c] = tile[r][c];
    }
}

// ---------------------------------------------------------------------------
// GEMM: C[M,N] = A[M,K] * Bt^T, Bt[N][K] pre-transposed f16.
// 8 waves; block tile 256x64; wave tile 32x64 (acc[2][4]).
// Bt tile double-buffered in LDS via async copy; one barrier per K-step.
// ---------------------------------------------------------------------------
template <typename AT, bool OUT_HALF>
__global__ __launch_bounds__(256) void gemm_wmma_kernel(
    const AT* __restrict__ A, const _Float16* __restrict__ Bt, void* __restrict__ Cm,
    int M, int N, int K, float out_scale)
{
    __shared__ __align__(16) _Float16 sBt[2][64][40];   // [buf][n][k], 20-dword stride

    const int tid  = threadIdx.x;
    const int wave = tid >> 5;
    const int lane = tid & 31;
    const int l16  = lane & 15;
    const int hh   = lane >> 4;

    const int m0 = blockIdx.y * 256 + wave * 32;
    const int n0 = blockIdx.x * 64;

    const int sn = tid >> 2;            // staging row (n) 0..63
    const int sk = (tid & 3) * 8;       // staging k offset

    auto load_af = [&](const AT* arow) -> v16h {
        if constexpr (sizeof(AT) == 2) {
            v8h lo = *(const v8h*)(arow + 8 * hh);
            v8h hi = *(const v8h*)(arow + 16 + 8 * hh);
            return combine(lo, hi);
        } else {
            const f4* p0 = (const f4*)(arow + 8 * hh);
            const f4* p1 = (const f4*)(arow + 16 + 8 * hh);
            return combine(cvt_v8h(p0[0], p0[1]), cvt_v8h(p1[0], p1[1]));
        }
    };

    v8f acc[2][4] = {};

    const _Float16* btrow = &Bt[(size_t)(n0 + sn) * K + sk];
    async_cp16(btrow, &sBt[0][sn][sk]);     // prologue: stage buffer 0

    int buf = 0;
    for (int k0 = 0; k0 < K; k0 += 32, buf ^= 1) {
        async_wait0();
        __syncthreads();
        if (k0 + 32 < K)
            async_cp16(btrow + k0 + 32, &sBt[buf ^ 1][sn][sk]);

        v16h af0 = load_af(A + (size_t)(m0 + l16) * K + k0);
        v16h af1 = load_af(A + (size_t)(m0 + 16 + l16) * K + k0);

#pragma unroll
        for (int t = 0; t < 4; ++t) {
            v8h b0 = *(const v8h*)&sBt[buf][t * 16 + l16][8 * hh];
            v8h b1 = *(const v8h*)&sBt[buf][t * 16 + l16][16 + 8 * hh];
            v16h bfrag = combine(b0, b1);
            acc[0][t] = wmma_f16(af0, bfrag, acc[0][t]);
            acc[1][t] = wmma_f16(af1, bfrag, acc[1][t]);
        }
    }

    // C/D layout: VGPR r -> row r + 8*hh; col = lane%16.
#pragma unroll
    for (int g = 0; g < 2; ++g) {
#pragma unroll
        for (int t = 0; t < 4; ++t) {
#pragma unroll
            for (int r = 0; r < 8; ++r) {
                size_t row = (size_t)(m0 + g * 16 + r + (hh << 3));
                size_t col = (size_t)(n0 + t * 16 + l16);
                float val = acc[g][t][r] * out_scale;
                if constexpr (OUT_HALF)
                    ((_Float16*)Cm)[row * N + col] = (_Float16)val;
                else
                    ((float*)Cm)[row * N + col] = val;
            }
        }
    }
}

// ---------------------------------------------------------------------------
// Flash attention: grid (S/128, HEADS, BATCH), 8 waves; wave = 16 query rows x
// full HEAD_DIM.  K/V tiles double-buffered via async copy; V pre-transposed.
// Q is pre-scaled by 1/sqrt(Dh) in the projection GEMM.
// ---------------------------------------------------------------------------
__global__ __launch_bounds__(256) void attn_wmma_kernel(
    const _Float16* __restrict__ Qp, const _Float16* __restrict__ Kp,
    const _Float16* __restrict__ Vt, const int* __restrict__ mask,
    _Float16* __restrict__ ctx)
{
    __shared__ __align__(16) _Float16 sK[2][32][HEAD_DIM + 8];   // [buf][key][d]
    __shared__ __align__(16) _Float16 sVt[2][HEAD_DIM][40];      // [buf][d][key]
    __shared__ __align__(16) _Float16 sP[8][16][40];             // per-wave P

    const int tid  = threadIdx.x;
    const int wave = tid >> 5;
    const int lane = tid & 31;
    const int l16  = lane & 15;
    const int hh   = lane >> 4;

    const int b  = blockIdx.z;
    const int h  = blockIdx.y;
    const int q0 = blockIdx.x * 128 + wave * 16;

    const size_t headOff = (size_t)h * HEAD_DIM;
    const size_t vtbase  = ((size_t)b * HEADS + h) * HEAD_DIM * SEQ;

    auto stage = [&](int buf, int kb) {
#pragma unroll
        for (int t = 0; t < 2; ++t) {
            int cc = tid + t * 256;             // 0..511 chunks of 8
            int kr = cc >> 4, dofs = (cc & 15) * 8;
            async_cp16(&Kp[((size_t)b * SEQ + kb + kr) * D_MODEL + headOff + dofs],
                       &sK[buf][kr][dofs]);
        }
#pragma unroll
        for (int t = 0; t < 2; ++t) {
            int cc = tid + t * 256;
            int d = cc >> 2, ko = (cc & 3) * 8;
            async_cp16(&Vt[vtbase + (size_t)d * SEQ + kb + ko],
                       &sVt[buf][d][ko]);
        }
    };

    // Q fragments: 4 frags covering d = 0..127 (b128 loads)
    v16h qf[4];
    {
        const _Float16* qrow =
            Qp + ((size_t)b * SEQ + q0 + l16) * D_MODEL + headOff;
#pragma unroll
        for (int f = 0; f < 4; ++f) {
            v8h lo = *(const v8h*)(qrow + f * 32 + 8 * hh);
            v8h hi = *(const v8h*)(qrow + f * 32 + 16 + 8 * hh);
            qf[f] = combine(lo, hi);
        }
    }

    v8f   acc[8] = {};
    float mrow[8], lrow[8];
#pragma unroll
    for (int r = 0; r < 8; ++r) { mrow[r] = -3.0e38f; lrow[r] = 0.0f; }

    stage(0, 0);                                 // prologue

    int buf = 0;
    for (int kb = 0; kb < SEQ; kb += 32, buf ^= 1) {
        async_wait0();
        __syncthreads();
        if (kb + 32 < SEQ) stage(buf ^ 1, kb + 32);

        // scores S = Q * K^T : two 16x16 frags (keys kb..+15 / kb+16..+31)
        v8f s0 = {}, s1 = {};
#pragma unroll
        for (int f = 0; f < 4; ++f) {
            v8h a0 = *(const v8h*)&sK[buf][l16][f * 32 + 8 * hh];
            v8h a1 = *(const v8h*)&sK[buf][l16][f * 32 + 16 + 8 * hh];
            v8h c0 = *(const v8h*)&sK[buf][16 + l16][f * 32 + 8 * hh];
            v8h c1 = *(const v8h*)&sK[buf][16 + l16][f * 32 + 16 + 8 * hh];
            s0 = wmma_f16(qf[f], combine(a0, a1), s0);
            s1 = wmma_f16(qf[f], combine(c0, c1), s1);
        }

        // mask + online softmax (row r + 8*hh lives in one DPP row)
#pragma unroll
        for (int r = 0; r < 8; ++r) {
            int qg = q0 + r + (hh << 3);
            const int* mp = mask + (size_t)qg * SEQ + kb;
            float v0 = s0[r];
            float v1 = s1[r];
            if (mp[l16] == 0)      v0 = -1.0e9f;
            if (mp[16 + l16] == 0) v1 = -1.0e9f;

            float mx   = row_max16(fmaxf(v0, v1));
            float mnew = fmaxf(mrow[r], mx);
            float corr = __expf(mrow[r] - mnew);
            float p0   = __expf(v0 - mnew);
            float p1   = __expf(v1 - mnew);
            s0[r] = p0; s1[r] = p1;

            float rs = row_sum16(p0 + p1);
            lrow[r] = lrow[r] * corr + rs;
            mrow[r] = mnew;
#pragma unroll
            for (int f = 0; f < 8; ++f) acc[f][r] *= corr;
        }

        // P: C-layout -> A-layout via per-wave LDS round trip (in-order per wave).
#pragma unroll
        for (int r = 0; r < 8; ++r) {
            sP[wave][r + (hh << 3)][l16]      = (_Float16)s0[r];
            sP[wave][r + (hh << 3)][16 + l16] = (_Float16)s1[r];
        }
        asm volatile("s_wait_dscnt 0" ::: "memory");
        v8h plo = *(const v8h*)&sP[wave][l16][8 * hh];
        v8h phi = *(const v8h*)&sP[wave][l16][16 + 8 * hh];
        v16h pf = combine(plo, phi);

        // O += P * V : contraction over 32 keys, 8 frags across d (b128 reads).
#pragma unroll
        for (int f = 0; f < 8; ++f) {
            v8h b0 = *(const v8h*)&sVt[buf][f * 16 + l16][8 * hh];
            v8h b1 = *(const v8h*)&sVt[buf][f * 16 + l16][16 + 8 * hh];
            acc[f] = wmma_f16(pf, combine(b0, b1), acc[f]);
        }
        // no trailing barrier: next iteration's wait+barrier guards buffer reuse
    }

    // finalize: divide by l, write merged-head context [B, S, D] as f16.
#pragma unroll
    for (int f = 0; f < 8; ++f) {
#pragma unroll
        for (int r = 0; r < 8; ++r) {
            int qg = q0 + r + (hh << 3);
            float o = acc[f][r] / lrow[r];
            ctx[((size_t)b * SEQ + qg) * D_MODEL + headOff + f * 16 + l16] =
                (_Float16)o;
        }
    }
}

// ---------------------------------------------------------------------------

extern "C" void kernel_launch(void* const* d_in, const int* in_sizes, int n_in,
                              void* d_out, int out_size, void* d_ws, size_t ws_size,
                              hipStream_t stream)
{
    const float* q    = (const float*)d_in[0];
    const float* k    = (const float*)d_in[1];
    const float* v    = (const float*)d_in[2];
    const int*   mask = (const int*)d_in[3];
    const float* Wq   = (const float*)d_in[4];
    const float* Wk   = (const float*)d_in[5];
    const float* Wv   = (const float*)d_in[6];
    const float* Wo   = (const float*)d_in[7];
    float*       out  = (float*)d_out;

    const size_t MD = (size_t)MROWS * D_MODEL;
    const size_t DD = (size_t)D_MODEL * D_MODEL;
    _Float16* Qp  = (_Float16*)d_ws;
    _Float16* Kp  = Qp  + MD;
    _Float16* Vp  = Kp  + MD;
    _Float16* ctx = Vp  + MD;
    _Float16* Vt  = ctx + MD;
    _Float16* Wqt = Vt  + MD;
    _Float16* Wkt = Wqt + DD;
    _Float16* Wvt = Wkt + DD;
    _Float16* Wot = Wvt + DD;

    dim3 thr(256);

    dim3 tgrid(D_MODEL / 32, D_MODEL / 32);       // (64, 64)
    wtrans_kernel<<<tgrid, thr, 0, stream>>>(Wq, Wqt, D_MODEL, D_MODEL);
    wtrans_kernel<<<tgrid, thr, 0, stream>>>(Wk, Wkt, D_MODEL, D_MODEL);
    wtrans_kernel<<<tgrid, thr, 0, stream>>>(Wv, Wvt, D_MODEL, D_MODEL);
    wtrans_kernel<<<tgrid, thr, 0, stream>>>(Wo, Wot, D_MODEL, D_MODEL);

    const float qscale = 0.08838834764831845f;    // 1/sqrt(HEAD_DIM)
    dim3 gblk(D_MODEL / 64, MROWS / 256);         // (32, 16)
    gemm_wmma_kernel<float, true><<<gblk, thr, 0, stream>>>(
        q, Wqt, Qp, MROWS, D_MODEL, D_MODEL, qscale);
    gemm_wmma_kernel<float, true><<<gblk, thr, 0, stream>>>(
        k, Wkt, Kp, MROWS, D_MODEL, D_MODEL, 1.0f);
    gemm_wmma_kernel<float, true><<<gblk, thr, 0, stream>>>(
        v, Wvt, Vp, MROWS, D_MODEL, D_MODEL, 1.0f);

    dim3 vgrid(SEQ / 32, D_MODEL / 32, BATCH);    // (64, 64, 2)
    vtrans_kernel<<<vgrid, thr, 0, stream>>>(Vp, Vt);

    dim3 ablk(SEQ / 128, HEADS, BATCH);           // (16, 16, 2)
    attn_wmma_kernel<<<ablk, thr, 0, stream>>>(Qp, Kp, Vt, mask, ctx);

    gemm_wmma_kernel<_Float16, false><<<gblk, thr, 0, stream>>>(
        ctx, Wot, out, MROWS, D_MODEL, D_MODEL, 1.0f);
}
